// LinformerAttention_21509196218784
// MI455X (gfx1250) — matrix operand order repaired and checked by reading, not compiled, not run on hardware
//
#include <hip/hip_runtime.h>
#include <hip/hip_bf16.h>

// ---- CDNA5 WMMA types ----
typedef __attribute__((ext_vector_type(8)))  __bf16 v8bf;
typedef __attribute__((ext_vector_type(16))) __bf16 v16bf;
typedef __attribute__((ext_vector_type(8)))  float  v8f;
typedef __attribute__((ext_vector_type(4)))  int    v4i;

#define WMMA_BF16(a, b, c) \
  __builtin_amdgcn_wmma_f32_16x16x32_bf16(false, (a), false, (b), (short)0, (c), false, false)

static constexpr int BSZ  = 4;
static constexpr int NSEQ = 4096;
static constexpr int CDIM = 1024;
static constexpr int HH   = 16;
static constexpr int KLM  = 256;
static constexpr int DH   = 64;
static constexpr int QKVLD = 3 * CDIM;

__device__ __forceinline__ v8f zero8() {
  v8f z;
#pragma unroll
  for (int i = 0; i < 8; ++i) z[i] = 0.0f;
  return z;
}

// ---------- 16-byte global->LDS copy, async if the toolchain has it ----------
__device__ __forceinline__ void copy16B_g2l(__bf16* lp, const __bf16* gp) {
#if __has_builtin(__builtin_amdgcn_global_load_async_to_lds_b128)
  typedef __attribute__((address_space(1))) v4i gv4i;
  typedef __attribute__((address_space(3))) v4i lv4i;
  __builtin_amdgcn_global_load_async_to_lds_b128(
      (gv4i*)(v4i*)(gp), (lv4i*)(v4i*)(lp), 0, 0);
#else
  *(v8bf*)lp = *(const v8bf*)gp;
#endif
}
__device__ __forceinline__ void wait_async_copies() {
#if __has_builtin(__builtin_amdgcn_global_load_async_to_lds_b128)
#if __has_builtin(__builtin_amdgcn_s_wait_asynccnt)
  __builtin_amdgcn_s_wait_asynccnt(0);
#else
  asm volatile("s_wait_asynccnt 0" ::: "memory");
#endif
#endif
}

// ---------- fragment loaders (bf16 sources, generic ptr: global or LDS) ----------
// A 16x32: lane L -> row m0+(L&15); K-runs of 8 at k0+((L>>4)<<3) and +16.
__device__ __forceinline__ v16bf fragA(const __bf16* A, long lda, int m0, int k0, int lane) {
  const __bf16* p = A + (long)(m0 + (lane & 15)) * lda + k0 + ((lane >> 4) << 3);
  v8bf lo = *(const v8bf*)p;
  v8bf hi = *(const v8bf*)(p + 16);
  return __builtin_shufflevector(lo, hi, 0,1,2,3,4,5,6,7,8,9,10,11,12,13,14,15);
}
// B 32x16 where column n of B is contiguous in memory (W rows = B columns):
// lane L -> col n0+(L&15); K-runs of 8 at k0+((L>>4)<<3) and +16.
__device__ __forceinline__ v16bf fragB(const __bf16* W, long ldw, int n0, int k0, int lane) {
  const __bf16* p = W + (long)(n0 + (lane & 15)) * ldw + k0 + ((lane >> 4) << 3);
  v8bf lo = *(const v8bf*)p;
  v8bf hi = *(const v8bf*)(p + 16);
  return __builtin_shufflevector(lo, hi, 0,1,2,3,4,5,6,7,8,9,10,11,12,13,14,15);
}

// ---------- C-tile stores ----------
__device__ __forceinline__ void storeC_f32(float* O, long ldo, int m0, int n0, int lane, v8f acc) {
  float* p = O + (long)(m0 + ((lane >> 4) << 3)) * ldo + n0 + (lane & 15);
#pragma unroll
  for (int v = 0; v < 8; ++v) p[(long)v * ldo] = acc[v];
}
__device__ __forceinline__ void storeC_bf16(__bf16* O, long ldo, int m0, int n0, int lane, v8f acc) {
  __bf16* p = O + (long)(m0 + ((lane >> 4) << 3)) * ldo + n0 + (lane & 15);
#pragma unroll
  for (int v = 0; v < 8; ++v) p[(long)v * ldo] = (__bf16)acc[v];
}
// Transposed store: element (m,n) -> O[n*ldo + m]; per lane n fixed, m contiguous -> 16B store.
__device__ __forceinline__ void storeC_T_bf16(__bf16* O, long ldo, int m0, int n0, int lane, v8f acc) {
  __bf16* p = O + (long)(n0 + (lane & 15)) * ldo + m0 + ((lane >> 4) << 3);
  v8bf v;
#pragma unroll
  for (int i = 0; i < 8; ++i) v[i] = (__bf16)acc[i];
  *(v8bf*)p = v;
}

// ------------------- Kernel 0: f32 -> bf16 convert -------------------
__global__ void __launch_bounds__(256) k_cvt(const float* __restrict__ src,
                                             __bf16* __restrict__ dst, long n) {
  long i = ((long)blockIdx.x * 256 + threadIdx.x) * 8;
  if (i >= n) return;
  float4 a = *(const float4*)(src + i);
  float4 b = *(const float4*)(src + i + 4);
  v8bf o;
  o[0]=(__bf16)a.x; o[1]=(__bf16)a.y; o[2]=(__bf16)a.z; o[3]=(__bf16)a.w;
  o[4]=(__bf16)b.x; o[5]=(__bf16)b.y; o[6]=(__bf16)b.z; o[7]=(__bf16)b.w;
  *(v8bf*)(dst + i) = o;
}

// ------------------- Kernel 1: QKV GEMM -------------------
// xb[16384,1024] @ wqb[3072,1024]^T. Q -> qb[16384,1024] row-major;
// K,V -> ktb/vtb[bh][64][4096] TRANSPOSED (d-major) for later contiguous B-frags.
__global__ void __launch_bounds__(256) k_qkv(const __bf16* __restrict__ xb,
                                             const __bf16* __restrict__ wqb,
                                             __bf16* __restrict__ qb,
                                             __bf16* __restrict__ ktb,
                                             __bf16* __restrict__ vtb) {
  int wid  = (blockIdx.x * 256 + threadIdx.x) >> 5;
  int lane = threadIdx.x & 31;
  const int ntn = QKVLD / 64;            // 48
  int mt = wid / ntn, nt = wid % ntn;
  int m0 = mt * 16, n0 = nt * 64;        // n0 stays inside one (s,h) block

  v8f acc[4];
#pragma unroll
  for (int j = 0; j < 4; ++j) acc[j] = zero8();

  for (int k0 = 0; k0 < CDIM; k0 += 32) {
    v16bf a = fragA(xb, CDIM, m0, k0, lane);
    if (k0 + 32 < CDIM)
      __builtin_prefetch(xb + (long)(m0 + (lane & 15)) * CDIM + k0 + 32, 0, 1);
#pragma unroll
    for (int j = 0; j < 4; ++j) {
      v16bf b = fragB(wqb, CDIM, n0 + j * 16, k0, lane);
      acc[j] = WMMA_BF16(a, b, acc[j]);
    }
  }

  int s  = n0 >> 10;                 // 0:q 1:k 2:v  (same for all 4 sub-tiles)
  int h  = (n0 & 1023) >> 6;
  int b_ = m0 >> 12;                 // batch
  int nb = m0 & 4095;                // seq row within batch
  long bh = (long)b_ * HH + h;
#pragma unroll
  for (int j = 0; j < 4; ++j) {
    int d0 = (n0 & 63) + j * 16;     // 0..48
    if (s == 0) {
      storeC_bf16(qb, CDIM, m0, h * DH + d0, lane, acc[j]);
    } else {
      __bf16* T = (s == 1 ? ktb : vtb) + bh * DH * NSEQ;   // [64][4096]
      storeC_T_bf16(T, NSEQ, nb, d0, lane, acc[j]);
    }
  }
}

// ------------------- Kernel 2: landmarks -------------------
// k_lm/v_lm = E[h][256,4096] @ {Kt,Vt}[bh][64,4096]^T (contiguous B-frags).
// k_lm -> klmb[bh][256][64]; v_lm -> vlmtb[bh][64][256] (transposed).
__global__ void __launch_bounds__(256) k_lmproj(const __bf16* __restrict__ eb,
                                                const __bf16* __restrict__ ktb,
                                                const __bf16* __restrict__ vtb,
                                                __bf16* __restrict__ klmb,
                                                __bf16* __restrict__ vlmtb) {
  int wid  = (blockIdx.x * 256 + threadIdx.x) >> 5;
  int lane = threadIdx.x & 31;
  int bh  = wid >> 5;                   // 0..63
  int rem = wid & 31;
  int mat = rem >> 4;                   // 0 -> k_lm, 1 -> v_lm
  int mt  = rem & 15;
  int h = bh & 15;
  int m0 = mt * 16;

  const __bf16* A = eb + (long)h * KLM * NSEQ;                     // [256,4096]
  const __bf16* W = (mat ? vtb : ktb) + (long)bh * DH * NSEQ;      // [64,4096]

  v8f acc[4];
#pragma unroll
  for (int j = 0; j < 4; ++j) acc[j] = zero8();

  for (int k0 = 0; k0 < NSEQ; k0 += 32) {
    v16bf a = fragA(A, NSEQ, m0, k0, lane);
#pragma unroll
    for (int j = 0; j < 4; ++j) {
      v16bf b = fragB(W, NSEQ, j * 16, k0, lane);
      acc[j] = WMMA_BF16(a, b, acc[j]);
    }
  }
  if (mat == 0) {
    __bf16* O = klmb + (long)bh * KLM * DH;        // [256][64]
#pragma unroll
    for (int j = 0; j < 4; ++j) storeC_bf16(O, DH, m0, j * 16, lane, acc[j]);
  } else {
    __bf16* O = vlmtb + (long)bh * DH * KLM;       // [64][256] transposed
#pragma unroll
    for (int j = 0; j < 4; ++j) storeC_T_bf16(O, KLM, m0, j * 16, lane, acc[j]);
  }
}

// ------------------- Kernel 3: fused landmark attention -------------------
// Block = 4 waves sharing one (b,h); k_lm panel async-staged into LDS.
__global__ void __launch_bounds__(128) k_attn(const __bf16* __restrict__ qb,
                                              const __bf16* __restrict__ klmb,
                                              const __bf16* __restrict__ vlmtb,
                                              __bf16* __restrict__ aoutb) {
  __shared__ __bf16 sKL[KLM * DH];        // 32 KB: k_lm [256][64], shared by block
  __shared__ __bf16 sP[4][16 * KLM];      // 32 KB: probs per wave
  int wave = threadIdx.x >> 5, lane = threadIdx.x & 31;
  int wid = blockIdx.x * 4 + wave;        // 4 consecutive wids share bh (256 | tiles)
  int nt = wid & 255;
  int bh = wid >> 8;
  int b_ = bh >> 4, h = bh & 15;
  int n0 = nt * 16;

  // ---- stage k_lm panel: 2048 x 16B chunks over 128 threads ----
  {
    const __bf16* KLg = klmb + (long)bh * KLM * DH;
#pragma unroll
    for (int i = 0; i < 16; ++i) {
      int c = threadIdx.x + i * 128;            // chunk id
      copy16B_g2l(sKL + c * 8, KLg + c * 8);
    }
    wait_async_copies();
    __syncthreads();
  }

  // ---- scores: Q(16x64) @ k_lm^T -> 16 tiles over 256 landmarks ----
  const __bf16* Q = qb;                   // row = b*4096+n, col = h*64+d
  v8f s[16];
#pragma unroll
  for (int t = 0; t < 16; ++t) s[t] = zero8();
  for (int k0 = 0; k0 < DH; k0 += 32) {
    v16bf a = fragA(Q, CDIM, b_ * NSEQ + n0, h * DH + k0, lane);
#pragma unroll
    for (int t = 0; t < 16; ++t) {
      v16bf bf = fragB(sKL, DH, t * 16, k0, lane);   // from LDS
      s[t] = WMMA_BF16(a, bf, s[t]);
    }
  }

  // ---- softmax over landmarks (scale folded into exp) ----
  const float scale = 0.125f;
  float rmax[8], rsum[8];
#pragma unroll
  for (int v = 0; v < 8; ++v) rmax[v] = -3.4e38f;
#pragma unroll
  for (int t = 0; t < 16; ++t)
#pragma unroll
    for (int v = 0; v < 8; ++v) rmax[v] = fmaxf(rmax[v], s[t][v]);
#pragma unroll
  for (int m = 8; m >= 1; m >>= 1)
#pragma unroll
    for (int v = 0; v < 8; ++v) rmax[v] = fmaxf(rmax[v], __shfl_xor(rmax[v], m, 32));
#pragma unroll
  for (int v = 0; v < 8; ++v) rsum[v] = 0.0f;
#pragma unroll
  for (int t = 0; t < 16; ++t)
#pragma unroll
    for (int v = 0; v < 8; ++v) {
      float e = __expf(scale * (s[t][v] - rmax[v]));
      s[t][v] = e;
      rsum[v] += e;
    }
#pragma unroll
  for (int m = 8; m >= 1; m >>= 1)
#pragma unroll
    for (int v = 0; v < 8; ++v) rsum[v] += __shfl_xor(rsum[v], m, 32);
#pragma unroll
  for (int v = 0; v < 8; ++v) rsum[v] = 1.0f / rsum[v];

  // ---- probs -> per-wave LDS as bf16 [16][256] ----
  __bf16* S = sP[wave];
  int rbase = (lane >> 4) << 3;
  int col = lane & 15;
#pragma unroll
  for (int t = 0; t < 16; ++t)
#pragma unroll
    for (int v = 0; v < 8; ++v)
      S[(rbase + v) * KLM + t * 16 + col] = (__bf16)(s[t][v] * rsum[v]);

  // ---- out = P(16x256) @ v_lm(256x64); B-frags contiguous from vlmtb ----
  const __bf16* VT = vlmtb + (long)bh * DH * KLM;   // [64][256]
  v8f o[4];
#pragma unroll
  for (int j = 0; j < 4; ++j) o[j] = zero8();
  for (int k0 = 0; k0 < KLM; k0 += 32) {
    v16bf a = fragA(S, KLM, 0, k0, lane);
#pragma unroll
    for (int j = 0; j < 4; ++j) {
      v16bf bf = fragB(VT, KLM, j * 16, k0, lane);
      o[j] = WMMA_BF16(a, bf, o[j]);
    }
  }
  // flat (b,h,n,d) order == reference's reshape
  __bf16* O = aoutb + (long)bh * NSEQ * DH;
#pragma unroll
  for (int j = 0; j < 4; ++j) storeC_bf16(O, DH, n0, j * 16, lane, o[j]);
}

// ------------------- Kernel 4: out = A @ Wproj^T + bproj (f32 out) ----------
__global__ void __launch_bounds__(256) k_proj(const __bf16* __restrict__ Ain,
                                              const __bf16* __restrict__ wpb,
                                              const float* __restrict__ bias,
                                              float* __restrict__ out) {
  int wid  = (blockIdx.x * 256 + threadIdx.x) >> 5;
  int lane = threadIdx.x & 31;
  const int ntn = CDIM / 64;             // 16
  int mt = wid / ntn, nt = wid % ntn;
  int m0 = mt * 16, n0 = nt * 64;

  v8f acc[4];
#pragma unroll
  for (int j = 0; j < 4; ++j) acc[j] = zero8();

  for (int k0 = 0; k0 < CDIM; k0 += 32) {
    v16bf a = fragA(Ain, CDIM, m0, k0, lane);
    if (k0 + 32 < CDIM)
      __builtin_prefetch(Ain + (long)(m0 + (lane & 15)) * CDIM + k0 + 32, 0, 1);
#pragma unroll
    for (int j = 0; j < 4; ++j) {
      v16bf b = fragB(wpb, CDIM, n0 + j * 16, k0, lane);
      acc[j] = WMMA_BF16(a, b, acc[j]);
    }
  }
#pragma unroll
  for (int j = 0; j < 4; ++j) {
    float bv = bias[n0 + j * 16 + (lane & 15)];
#pragma unroll
    for (int v = 0; v < 8; ++v) acc[j][v] += bv;
    storeC_f32(out, CDIM, m0, n0 + j * 16, lane, acc[j]);
  }
}

// ------------------------------- launcher -------------------------------
extern "C" void kernel_launch(void* const* d_in, const int* in_sizes, int n_in,
                              void* d_out, int out_size, void* d_ws, size_t ws_size,
                              hipStream_t stream) {
  const float* x     = (const float*)d_in[0];
  const float* Wqkv  = (const float*)d_in[1];
  const float* E     = (const float*)d_in[2];
  const float* Wproj = (const float*)d_in[3];
  const float* bproj = (const float*)d_in[4];
  float* out = (float*)d_out;

  const long NX = (long)BSZ * NSEQ * CDIM;       // 16,777,216
  const long NWQ = (long)QKVLD * CDIM;           //  3,145,728
  const long NE  = (long)HH * KLM * NSEQ;        // 16,777,216
  const long NWP = (long)CDIM * CDIM;            //  1,048,576
  const long NKT = (long)BSZ * HH * DH * NSEQ;   // 16,777,216
  const long NKLM = (long)BSZ * HH * KLM * DH;   //  1,048,576

  __bf16* p = (__bf16*)d_ws;
  __bf16* xb    = p; p += NX;
  __bf16* wqb   = p; p += NWQ;
  __bf16* eb    = p; p += NE;
  __bf16* wpb   = p; p += NWP;
  __bf16* qb    = p; p += NX;     // [16384][1024]
  __bf16* ktb   = p; p += NKT;    // [64][64][4096]
  __bf16* vtb   = p; p += NKT;
  __bf16* klmb  = p; p += NKLM;   // [64][256][64]
  __bf16* vlmtb = p; p += NKLM;   // [64][64][256]
  __bf16* aoutb = p; p += NX;     // [16384][1024]

  // converts (vector 8 elems/thread)
  k_cvt<<<(int)(NX  / (8 * 256)), 256, 0, stream>>>(x, xb, NX);
  k_cvt<<<(int)(NWQ / (8 * 256)), 256, 0, stream>>>(Wqkv, wqb, NWQ);
  k_cvt<<<(int)(NE  / (8 * 256)), 256, 0, stream>>>(E, eb, NE);
  k_cvt<<<(int)(NWP / (8 * 256)), 256, 0, stream>>>(Wproj, wpb, NWP);

  k_qkv<<<6144, 256, 0, stream>>>(xb, wqb, qb, ktb, vtb);       // 49152 waves
  k_lmproj<<<256, 256, 0, stream>>>(eb, ktb, vtb, klmb, vlmtb); //  2048 waves
  k_attn<<<4096, 128, 0, stream>>>(qb, klmb, vlmtb, aoutb);     // 16384 waves
  k_proj<<<2048, 256, 0, stream>>>(aoutb, wpb, bproj, out);     // 16384 waves
}
